// SimpleMoE_1812476199415
// MI455X (gfx1250) — compile-verified, matrix operand order checked by reading, not run on hardware
//
#include <hip/hip_runtime.h>
#include <hip/hip_bf16.h>
#include <stdint.h>

// Problem constants (match reference)
#define NB 2
#define NT 2048
#define ND 1024
#define NF 4096
#define NE 8
#define NTOK (NB * NT)        // 4096 tokens
#define NSLOT (NTOK * 2)      // 8192 (token, slot) pairs

#define TM 128                // token rows per workgroup tile
#define KC 32                 // K chunk (bf16 WMMA K)
#define MAX_MT (NTOK / TM)    // 32 worst-case M tiles per expert
#define ASTRIDE 40            // padded LDS row stride in halves (80B)

typedef __bf16 bf16_t;
typedef __bf16 v16bf __attribute__((ext_vector_type(16)));
typedef __bf16 v8bf  __attribute__((ext_vector_type(8)));
typedef float  v8f   __attribute__((ext_vector_type(8)));
typedef float  v4f   __attribute__((ext_vector_type(4)));

// ---------------------------------------------------------------------------
// Async global->LDS stage (CDNA5 GLOBAL_LOAD_ASYNC_TO_LDS_B128, ASYNCcnt).
// GVS addressing: mem = SADDR64 + VADDR32 (+inst offset). VDST carries the
// per-lane LDS byte address (low 32 bits of a generic LDS pointer).
// ---------------------------------------------------------------------------
__device__ __forceinline__ void async_b128(unsigned lds_addr, unsigned goff,
                                           unsigned long long base)
{
    asm volatile("global_load_async_to_lds_b128 %0, %1, %2"
                 :: "v"(lds_addr), "v"(goff), "s"(base)
                 : "memory");
}
__device__ __forceinline__ void wait_async0()
{
    asm volatile("s_wait_asynccnt 0x0" ::: "memory");
}

// B-fragment prefetch registers (f32) and conversion to a 32x16 bf16 fragment.
// Lane L: column n=L&15, K halfset koff=(L>>4)*8 per ISA 7.12.2:
//   frag[0..7] = W[row][kb+koff+0..7], frag[8..15] = W[row][kb+koff+16..23]
struct BPF { v4f q[4]; };
__device__ __forceinline__ BPF load_b_regs(const float* bp)
{
    BPF r;
    r.q[0] = *(const v4f*)(bp + 0);
    r.q[1] = *(const v4f*)(bp + 4);
    r.q[2] = *(const v4f*)(bp + 16);
    r.q[3] = *(const v4f*)(bp + 20);
    return r;
}
__device__ __forceinline__ v16bf cvt_b(const BPF& r)
{
    v16bf b;
#pragma unroll
    for (int j = 0; j < 4; ++j) {
        b[j]      = (bf16_t)r.q[0][j];
        b[4 + j]  = (bf16_t)r.q[1][j];
        b[8 + j]  = (bf16_t)r.q[2][j];
        b[12 + j] = (bf16_t)r.q[3][j];
    }
    return b;
}

// ---------------------------------------------------------------------------
// 0) One-shot x f32 -> bf16 (enables pure-copy async A staging in GEMM1).
// ---------------------------------------------------------------------------
__global__ __launch_bounds__(256) void moe_cvt_x(
    const float* __restrict__ x, bf16_t* __restrict__ xb)
{
    const int i = (blockIdx.x * 256 + threadIdx.x) * 8;
    v4f a = *(const v4f*)(x + i);
    v4f b = *(const v4f*)(x + i + 4);
    v8bf r;
#pragma unroll
    for (int j = 0; j < 4; ++j) { r[j] = (bf16_t)a[j]; r[4 + j] = (bf16_t)b[j]; }
    *(v8bf*)(xb + i) = r;
}

// ---------------------------------------------------------------------------
// 1) Router: scores = x @ Wr^T, per-token top-2 (first-index tie break).
// ---------------------------------------------------------------------------
__global__ __launch_bounds__(256) void moe_router_topk(
    const float* __restrict__ x, const float* __restrict__ Wr,
    float* __restrict__ topk_val, int* __restrict__ topk_idx)
{
    __shared__ float sWr[NE * ND];
    for (int i = threadIdx.x; i < NE * ND; i += 256) sWr[i] = Wr[i];
    __syncthreads();

    const int wave = threadIdx.x >> 5;
    const int lane = threadIdx.x & 31;
    const int tok  = blockIdx.x * 8 + wave;
    const float* xr = x + (size_t)tok * ND;

    float acc[NE];
#pragma unroll
    for (int e = 0; e < NE; ++e) acc[e] = 0.0f;

    for (int d = lane; d < ND; d += 32) {
        const float xv = xr[d];
#pragma unroll
        for (int e = 0; e < NE; ++e) acc[e] += xv * sWr[e * ND + d];
    }
#pragma unroll
    for (int e = 0; e < NE; ++e) {
#pragma unroll
        for (int off = 16; off > 0; off >>= 1)
            acc[e] += __shfl_xor(acc[e], off, 32);
    }
    if (lane == 0) {
        int i0 = 0; float v0 = acc[0];
#pragma unroll
        for (int e = 1; e < NE; ++e) if (acc[e] > v0) { v0 = acc[e]; i0 = e; }
        int i1 = -1; float v1 = -3.4e38f;
#pragma unroll
        for (int e = 0; e < NE; ++e)
            if (e != i0 && acc[e] > v1) { v1 = acc[e]; i1 = e; }
        topk_val[tok * 2 + 0] = v0;  topk_val[tok * 2 + 1] = v1;
        topk_idx[tok * 2 + 0] = i0;  topk_idx[tok * 2 + 1] = i1;
    }
}

// ---------------------------------------------------------------------------
// 2) Softmax over the sequence axis (reference axis=1), one block per (b, k).
// ---------------------------------------------------------------------------
__global__ __launch_bounds__(256) void moe_softmax_seq(
    const float* __restrict__ topk_val, float* __restrict__ weightv)
{
    const int b = blockIdx.x >> 1;
    const int k = blockIdx.x & 1;
    __shared__ float red[256];

    float m = -3.4e38f;
    for (int t = threadIdx.x; t < NT; t += 256)
        m = fmaxf(m, topk_val[(b * NT + t) * 2 + k]);
    red[threadIdx.x] = m; __syncthreads();
    for (int s = 128; s > 0; s >>= 1) {
        if (threadIdx.x < s) red[threadIdx.x] = fmaxf(red[threadIdx.x], red[threadIdx.x + s]);
        __syncthreads();
    }
    m = red[0]; __syncthreads();

    float sum = 0.0f;
    for (int t = threadIdx.x; t < NT; t += 256)
        sum += __expf(topk_val[(b * NT + t) * 2 + k] - m);
    red[threadIdx.x] = sum; __syncthreads();
    for (int s = 128; s > 0; s >>= 1) {
        if (threadIdx.x < s) red[threadIdx.x] += red[threadIdx.x + s];
        __syncthreads();
    }
    const float inv = 1.0f / red[0];
    for (int t = threadIdx.x; t < NT; t += 256) {
        const int i = (b * NT + t) * 2 + k;
        weightv[i] = __expf(topk_val[i] - m) * inv;
    }
}

// ---------------------------------------------------------------------------
// 3) Deterministic compaction (single block, integer scans -> bit-exact).
// ---------------------------------------------------------------------------
__global__ __launch_bounds__(256) void moe_compact(
    const int* __restrict__ topk_idx, int* __restrict__ counts,
    int* __restrict__ offsets, int* __restrict__ pos, int* __restrict__ tok_list)
{
    __shared__ int scnt[NE];
    __shared__ int base[NE];
    __shared__ int scan[256];

    if (threadIdx.x < NE) scnt[threadIdx.x] = 0;
    __syncthreads();
    for (int i = threadIdx.x; i < NSLOT; i += 256)
        atomicAdd(&scnt[topk_idx[i]], 1);
    __syncthreads();
    if (threadIdx.x == 0) {
        int run = 0;
        for (int e = 0; e < NE; ++e) {
            base[e] = run;
            counts[e] = scnt[e];
            offsets[e] = run;
            run += scnt[e];
        }
    }
    __syncthreads();

    for (int c = 0; c < NSLOT / 256; ++c) {
        const int i = c * 256 + threadIdx.x;
        const int ei = topk_idx[i];
        int myPos = 0;
        for (int e = 0; e < NE; ++e) {
            const int flag = (ei == e) ? 1 : 0;
            scan[threadIdx.x] = flag; __syncthreads();
            int v = flag;
            for (int s = 1; s < 256; s <<= 1) {
                const int add = (threadIdx.x >= s) ? scan[threadIdx.x - s] : 0;
                __syncthreads();
                v += add;
                scan[threadIdx.x] = v;
                __syncthreads();
            }
            if (ei == e) myPos = base[e] + v - 1;   // global row (offset + rank)
            const int tot = scan[255];
            __syncthreads();
            if (threadIdx.x == 0) base[e] += tot;
            __syncthreads();
        }
        pos[i] = myPos;
        tok_list[myPos] = i >> 1;
    }
}

// ---------------------------------------------------------------------------
// 4) GEMM1: h[row,f] = relu(x[tok(row),:].W1[e][f,:] + b1[e][f])  (bf16 out)
//    128 gathered rows x 128 F cols per workgroup; async double-buffered A
//    staging, register-prefetched B, 8 back-to-back WMMAs per K chunk/wave.
// ---------------------------------------------------------------------------
__global__ __launch_bounds__(256) void moe_gemm1(
    const bf16_t* __restrict__ xb, const float* __restrict__ W1,
    const float* __restrict__ b1,
    const int* __restrict__ counts, const int* __restrict__ offsets,
    const int* __restrict__ tok_list, bf16_t* __restrict__ h)
{
    const int e  = blockIdx.y >> 5;
    const int mt = blockIdx.y & 31;
    const int cnt = counts[e];
    if (mt * TM >= cnt) return;
    const int rowbase = offsets[e] + mt * TM;

    __shared__ bf16_t sA[2][TM * ASTRIDE];
    __shared__ int    sTok[TM];

    const int tid = threadIdx.x;
    if (tid < TM)
        sTok[tid] = (mt * TM + tid < cnt) ? tok_list[rowbase + tid] : 0;
    __syncthreads();

    const int wave = tid >> 5;
    const int lane = tid & 31;
    const int fcol = blockIdx.x * 128 + wave * 16 + (lane & 15);
    const int koff = (lane >> 4) << 3;                    // 0 or 8
    const float* Wrow = W1 + ((size_t)e * NF + fcol) * ND;

    const int arow = tid & (TM - 1);
    const int aseg = tid >> 7;                            // 0/1
    const unsigned long long xbase = (unsigned long long)(uintptr_t)xb;
    const unsigned gsrc0 =
        ((unsigned)sTok[arow] * ND + (unsigned)aseg * 16) * (unsigned)sizeof(bf16_t);
    unsigned ldsd[2];
    ldsd[0] = (unsigned)(uintptr_t)&sA[0][arow * ASTRIDE + aseg * 16];
    ldsd[1] = (unsigned)(uintptr_t)&sA[1][arow * ASTRIDE + aseg * 16];

    v8f acc[8];
#pragma unroll
    for (int m = 0; m < 8; ++m)
#pragma unroll
        for (int j = 0; j < 8; ++j) acc[m][j] = 0.0f;

    BPF bregs = load_b_regs(Wrow + koff);
    async_b128(ldsd[0], gsrc0, xbase);
    async_b128(ldsd[0] + 16, gsrc0 + 16, xbase);

    const int KITER = ND / KC;
    for (int i = 0; i < KITER; ++i) {
        const int kc = i * KC;
        wait_async0();
        __syncthreads();
        if (i + 1 < KITER) {
            const unsigned off = gsrc0 + (unsigned)(kc + KC) * sizeof(bf16_t);
            async_b128(ldsd[(i + 1) & 1], off, xbase);
            async_b128(ldsd[(i + 1) & 1] + 16, off + 16, xbase);
        }
        v16bf bfrag = cvt_b(bregs);
        if (i + 1 < KITER) bregs = load_b_regs(Wrow + kc + KC + koff);

        v16bf af[8];
#pragma unroll
        for (int m = 0; m < 8; ++m) {
            const bf16_t* ap = &sA[i & 1][(m * 16 + (lane & 15)) * ASTRIDE + koff];
            v8bf lo = *(const v8bf*)ap;
            v8bf hi = *(const v8bf*)(ap + 16);
#pragma unroll
            for (int q = 0; q < 8; ++q) { af[m][q] = lo[q]; af[m][q + 8] = hi[q]; }
        }
#pragma unroll
        for (int m = 0; m < 8; ++m)
            acc[m] = __builtin_amdgcn_wmma_f32_16x16x32_bf16(
                false, af[m], false, bfrag, (short)0, acc[m], false, false);
    }

    // Epilogue: +bias, relu, bf16 store. C layout: lanes 0-15 -> M=j,
    // lanes 16-31 -> M=j+8, col n = lane&15.
    const float bias = b1[e * NF + fcol];
    const int rhalf = (lane < 16) ? 0 : 8;
    bf16_t* hout = h + (size_t)rowbase * NF + fcol;
    if (mt * TM + TM <= cnt) {            // fast path: full tile
#pragma unroll
        for (int m = 0; m < 8; ++m)
#pragma unroll
            for (int j = 0; j < 8; ++j) {
                float v = acc[m][j] + bias;
                v = v > 0.0f ? v : 0.0f;
                hout[(size_t)(m * 16 + rhalf + j) * NF] = (bf16_t)v;
            }
    } else {
#pragma unroll
        for (int m = 0; m < 8; ++m)
#pragma unroll
            for (int j = 0; j < 8; ++j) {
                const int r = m * 16 + rhalf + j;
                if (mt * TM + r < cnt) {
                    float v = acc[m][j] + bias;
                    v = v > 0.0f ? v : 0.0f;
                    hout[(size_t)r * NF] = (bf16_t)v;
                }
            }
    }
}

// ---------------------------------------------------------------------------
// 5) GEMM2: slot_out[row,d] = h[row,:].W2[e][d,:] + b2[e][d]   (f32 out)
// ---------------------------------------------------------------------------
__global__ __launch_bounds__(256) void moe_gemm2(
    const bf16_t* __restrict__ h, const float* __restrict__ W2,
    const float* __restrict__ b2,
    const int* __restrict__ counts, const int* __restrict__ offsets,
    float* __restrict__ slot_out)
{
    const int e  = blockIdx.y >> 5;
    const int mt = blockIdx.y & 31;
    const int cnt = counts[e];
    if (mt * TM >= cnt) return;
    const int rowbase = offsets[e] + mt * TM;

    __shared__ bf16_t sA[2][TM * ASTRIDE];

    const int tid  = threadIdx.x;
    const int wave = tid >> 5;
    const int lane = tid & 31;
    const int dcol = blockIdx.x * 128 + wave * 16 + (lane & 15);
    const int koff = (lane >> 4) << 3;
    const float* Wrow = W2 + ((size_t)e * ND + dcol) * NF;

    const int arow = tid & (TM - 1);
    const int aseg = tid >> 7;
    const int grow = (mt * TM + arow < cnt) ? (rowbase + arow) : rowbase;
    const unsigned long long hbase = (unsigned long long)(uintptr_t)h;
    const unsigned gsrc0 =
        ((unsigned)grow * NF + (unsigned)aseg * 16) * (unsigned)sizeof(bf16_t);
    unsigned ldsd[2];
    ldsd[0] = (unsigned)(uintptr_t)&sA[0][arow * ASTRIDE + aseg * 16];
    ldsd[1] = (unsigned)(uintptr_t)&sA[1][arow * ASTRIDE + aseg * 16];

    v8f acc[8];
#pragma unroll
    for (int m = 0; m < 8; ++m)
#pragma unroll
        for (int j = 0; j < 8; ++j) acc[m][j] = 0.0f;

    BPF bregs = load_b_regs(Wrow + koff);
    async_b128(ldsd[0], gsrc0, hbase);
    async_b128(ldsd[0] + 16, gsrc0 + 16, hbase);

    const int KITER = NF / KC;
    for (int i = 0; i < KITER; ++i) {
        const int kc = i * KC;
        wait_async0();
        __syncthreads();
        if (i + 1 < KITER) {
            const unsigned off = gsrc0 + (unsigned)(kc + KC) * sizeof(bf16_t);
            async_b128(ldsd[(i + 1) & 1], off, hbase);
            async_b128(ldsd[(i + 1) & 1] + 16, off + 16, hbase);
        }
        v16bf bfrag = cvt_b(bregs);
        if (i + 1 < KITER) bregs = load_b_regs(Wrow + kc + KC + koff);

        v16bf af[8];
#pragma unroll
        for (int m = 0; m < 8; ++m) {
            const bf16_t* ap = &sA[i & 1][(m * 16 + (lane & 15)) * ASTRIDE + koff];
            v8bf lo = *(const v8bf*)ap;
            v8bf hi = *(const v8bf*)(ap + 16);
#pragma unroll
            for (int q = 0; q < 8; ++q) { af[m][q] = lo[q]; af[m][q + 8] = hi[q]; }
        }
#pragma unroll
        for (int m = 0; m < 8; ++m)
            acc[m] = __builtin_amdgcn_wmma_f32_16x16x32_bf16(
                false, af[m], false, bfrag, (short)0, acc[m], false, false);
    }

    const float bias = b2[e * ND + dcol];
    const int rhalf = (lane < 16) ? 0 : 8;
    float* so = slot_out + (size_t)rowbase * ND + dcol;
    if (mt * TM + TM <= cnt) {            // fast path: full tile
#pragma unroll
        for (int m = 0; m < 8; ++m)
#pragma unroll
            for (int j = 0; j < 8; ++j)
                so[(size_t)(m * 16 + rhalf + j) * ND] = acc[m][j] + bias;
    } else {
#pragma unroll
        for (int m = 0; m < 8; ++m)
#pragma unroll
            for (int j = 0; j < 8; ++j) {
                const int r = m * 16 + rhalf + j;
                if (mt * TM + r < cnt)
                    so[(size_t)r * ND] = acc[m][j] + bias;
            }
    }
}

// ---------------------------------------------------------------------------
// 6) Weighted gather: out[tok] = w0*slot_out[row0] + w1*slot_out[row1]
// ---------------------------------------------------------------------------
__global__ __launch_bounds__(256) void moe_gather(
    const float* __restrict__ slot_out, const float* __restrict__ weightv,
    const int* __restrict__ pos, float* __restrict__ out)
{
    const int tok = blockIdx.x;
    const float w0 = weightv[tok * 2 + 0];
    const float w1 = weightv[tok * 2 + 1];
    const float* s0 = slot_out + (size_t)pos[tok * 2 + 0] * ND;
    const float* s1 = slot_out + (size_t)pos[tok * 2 + 1] * ND;
    float* o = out + (size_t)tok * ND;

    const int d = threadIdx.x * 4;          // 256 threads x 4 = 1024 = ND
    v4f a = *(const v4f*)(s0 + d);
    v4f b = *(const v4f*)(s1 + d);
    v4f r;
#pragma unroll
    for (int j = 0; j < 4; ++j) r[j] = w0 * a[j] + w1 * b[j];
    *(v4f*)(o + d) = r;
}

// ---------------------------------------------------------------------------
// Launch. Inputs: x, Wr, W1, b1, W2, b2, top_k (==2).
// Workspace (~112.2 MB):
//   [0x00000) topk_val f32*8192   [0x08000) topk_idx i32*8192
//   [0x10000) weightv  f32*8192   [0x18000) pos      i32*8192
//   [0x20000) counts   i32*8      [0x20100) offsets  i32*8
//   [0x20200) tok_list i32*8192
//   [0x30000) h        bf16*8192*4096 (64MB)
//   [+64MB  ) slot_out f32 *8192*1024 (32MB)
//   [+32MB  ) xb       bf16*4096*1024 (8MB... 16MB region reserved)
// ---------------------------------------------------------------------------
extern "C" void kernel_launch(void* const* d_in, const int* in_sizes, int n_in,
                              void* d_out, int out_size, void* d_ws, size_t ws_size,
                              hipStream_t stream)
{
    (void)in_sizes; (void)n_in; (void)out_size; (void)ws_size;

    const float* x  = (const float*)d_in[0];
    const float* Wr = (const float*)d_in[1];
    const float* W1 = (const float*)d_in[2];
    const float* b1 = (const float*)d_in[3];
    const float* W2 = (const float*)d_in[4];
    const float* b2 = (const float*)d_in[5];
    float* out = (float*)d_out;

    char* ws = (char*)d_ws;
    float*  topk_val = (float*)(ws + 0x00000);
    int*    topk_idx = (int*)  (ws + 0x08000);
    float*  weightv  = (float*)(ws + 0x10000);
    int*    pos      = (int*)  (ws + 0x18000);
    int*    counts   = (int*)  (ws + 0x20000);
    int*    offsets  = (int*)  (ws + 0x20100);
    int*    tok_list = (int*)  (ws + 0x20200);
    bf16_t* h        = (bf16_t*)(ws + 0x30000);
    float*  slot_out = (float*)(ws + 0x30000 + (size_t)NSLOT * NF * sizeof(bf16_t));
    bf16_t* xb       = (bf16_t*)((char*)slot_out + (size_t)NSLOT * ND * sizeof(float));

    moe_cvt_x<<<(NTOK * ND) / (256 * 8), 256, 0, stream>>>(x, xb);
    moe_router_topk<<<NTOK / 8, 256, 0, stream>>>(x, Wr, topk_val, topk_idx);
    moe_softmax_seq<<<NB * 2, 256, 0, stream>>>(topk_val, weightv);
    moe_compact<<<1, 256, 0, stream>>>(topk_idx, counts, offsets, pos, tok_list);
    moe_gemm1<<<dim3(NF / 128, NE * MAX_MT), 256, 0, stream>>>(
        xb, W1, b1, counts, offsets, tok_list, h);
    moe_gemm2<<<dim3(ND / 128, NE * MAX_MT), 256, 0, stream>>>(
        h, W2, b2, counts, offsets, slot_out);
    moe_gather<<<NTOK, 256, 0, stream>>>(slot_out, weightv, pos, out);
}
